// NormLocalAttention_35433480192028
// MI455X (gfx1250) — matrix operand order repaired and checked by reading, not compiled, not run on hardware
//
#include <hip/hip_runtime.h>
#include <math.h>
#include <stdint.h>

// Problem constants (B, L, E, H, C from the reference)
#define BB 4
#define LL 4096
#define EE 1024
#define HH 16
#define GG 64                 // L / C chunks
#define MM (LL * BB)          // 16384 token rows
#define LDT 40                // padded LDS row stride (ushorts) for GEMM tiles

typedef __attribute__((ext_vector_type(16))) __bf16        v16bf;
typedef __attribute__((ext_vector_type(8)))  float         v8f;
typedef __attribute__((ext_vector_type(4)))  unsigned int  u32x4;
typedef __attribute__((ext_vector_type(2)))  unsigned int  u32x2;
typedef __attribute__((ext_vector_type(4)))  float         f32x4;

__device__ __forceinline__ unsigned short f2bf(float f) {
  unsigned int u = __float_as_uint(f);
  u += 0x7FFFu + ((u >> 16) & 1u);           // round-to-nearest-even
  return (unsigned short)(u >> 16);
}
__device__ __forceinline__ unsigned pack2(float a, float b) {
  return (unsigned)f2bf(a) | ((unsigned)f2bf(b) << 16);
}

// CDNA5 async global->LDS copy (16 bytes / lane), tracked with ASYNCcnt.
__device__ __forceinline__ void async_ld16(unsigned lds, unsigned long long gaddr) {
  asm volatile("global_load_async_to_lds_b128 %0, %1, off"
               :: "v"(lds), "v"(gaddr) : "memory");
}
__device__ __forceinline__ void wait_async0() {
  asm volatile("s_wait_asynccnt 0x0" ::: "memory");
}
// Generic->LDS offset: flat LDS aperture keeps the DS offset in addr[31:0].
__device__ __forceinline__ unsigned lds_off(const void* p) {
  return (unsigned)(uintptr_t)p;
}

union FragBF { v16bf v; u32x4 u[2]; };

// A operand (16xK=32 bf16) from row-major LDS tile. Lane l -> row (l&15).
// ISA 16-bit A layout: VGPR0..3 = K[8*hi .. +7], VGPR4..7 = K[16+8*hi .. +7].
__device__ __forceinline__ v16bf load_frag_a(const unsigned short* base, int row0,
                                             int stride, int k0, int lane) {
  const int r = lane & 15, hi = lane >> 4;
  const unsigned short* p = base + (size_t)(row0 + r) * stride + k0 + 8 * hi;
  FragBF f;
  f.u[0] = *(const u32x4*)(p);
  f.u[1] = *(const u32x4*)(p + 16);
  return f.v;
}
// B operand: lane l -> column (l&15); effective B[k][n] = src[n][k].
// B layout matches A: VGPR0..3 = K[8*hi .. +7], VGPR4..7 = K[16+8*hi .. +7].
__device__ __forceinline__ v16bf load_frag_b(const unsigned short* base, int row0,
                                             int stride, int k0, int lane) {
  const int r = lane & 15, hi = lane >> 4;
  const unsigned short* p = base + (size_t)(row0 + r) * stride + k0 + 8 * hi;
  FragBF f;
  f.u[0] = *(const u32x4*)(p);
  f.u[1] = *(const u32x4*)(p + 16);
  return f.v;
}

// ---------------------------------------------------------------------------
// K0a: fp32 query (B,L,E) -> bf16 x (M=L*B rows, E), transpose folded in.
// ---------------------------------------------------------------------------
__global__ __launch_bounds__(256)
void convert_x_kernel(const float* __restrict__ query,
                      unsigned short* __restrict__ xb) {
  const int m = blockIdx.x, tid = threadIdx.x;
  const float* src = query + ((size_t)(m & 3) * LL + (size_t)(m >> 2)) * EE + tid * 4;
  f32x4 v = *(const f32x4*)src;
  u32x2 p = {pack2(v.x, v.y), pack2(v.z, v.w)};
  *(u32x2*)(xb + (size_t)m * EE + tid * 4) = p;
}

// K0b: fp32 weights -> bf16, all four matrices into one buffer.
__global__ __launch_bounds__(256)
void convert_w_kernel(const float* __restrict__ Wq, const float* __restrict__ Wk,
                      const float* __restrict__ Wv, const float* __restrict__ Wo,
                      unsigned short* __restrict__ wball) {
  const int r = blockIdx.x, which = blockIdx.y, tid = threadIdx.x;
  const float* W = (which == 0) ? Wq : (which == 1) ? Wk : (which == 2) ? Wv : Wo;
  f32x4 v = *(const f32x4*)(W + (size_t)r * EE + tid * 4);
  u32x2 p = {pack2(v.x, v.y), pack2(v.z, v.w)};
  *(u32x2*)(wball + ((size_t)which * EE + r) * EE + tid * 4) = p;
}

// ---------------------------------------------------------------------------
// K1: QKV projections, double-buffered async-to-LDS staging.
// out = (x @ W.T + b) [* HD^-0.5 for q], bf16 result in (L,B,E).
// grid = (M/128, E/128, 3), block = 256 (8 waves, 4x2 wave grid).
// ---------------------------------------------------------------------------
__global__ __launch_bounds__(256)
void qkv_proj_kernel(const unsigned short* __restrict__ xb,
                     const unsigned short* __restrict__ wball,
                     const float* __restrict__ bq, const float* __restrict__ bk,
                     const float* __restrict__ bv,
                     unsigned short* __restrict__ qo,
                     unsigned short* __restrict__ ko,
                     unsigned short* __restrict__ vo) {
  __shared__ __align__(16) unsigned short As[2][128 * LDT];
  __shared__ __align__(16) unsigned short Bs[2][128 * LDT];

  const int which = blockIdx.z;
  const unsigned short* wb = wball + (size_t)which * EE * EE;
  const float* bias = (which == 0) ? bq : (which == 1) ? bk : bv;
  unsigned short* out = (which == 0) ? qo : (which == 1) ? ko : vo;
  const float alpha = (which == 0) ? 0.125f : 1.0f;

  const int m0 = blockIdx.x * 128, n0 = blockIdx.y * 128;
  const int tid = threadIdx.x, lane = tid & 31, wave = tid >> 5;
  const int wm = wave & 3, wn = wave >> 2;

  // staging: thread -> (row = tid/2, 16-col segment); 2x 16B chunks per tile.
  const int srow = tid >> 1, sseg = (tid & 1) * 16;
  const unsigned long long abase =
      (unsigned long long)(uintptr_t)(xb + (size_t)(m0 + srow) * EE + sseg);
  const unsigned long long bbase =
      (unsigned long long)(uintptr_t)(wb + (size_t)(n0 + srow) * EE + sseg);
  const unsigned alds[2] = {lds_off(&As[0][srow * LDT + sseg]),
                            lds_off(&As[1][srow * LDT + sseg])};
  const unsigned blds[2] = {lds_off(&Bs[0][srow * LDT + sseg]),
                            lds_off(&Bs[1][srow * LDT + sseg])};

  auto issue = [&](int kt, int buf) {
    const unsigned long long ko64 = (unsigned long long)kt * 64ull;  // 32 bf16
    async_ld16(alds[buf], abase + ko64);
    async_ld16(alds[buf] + 32, abase + ko64 + 32);
    async_ld16(blds[buf], bbase + ko64);
    async_ld16(blds[buf] + 32, bbase + ko64 + 32);
  };

  v8f acc[2][4];
  const v8f vzero = {0.f, 0.f, 0.f, 0.f, 0.f, 0.f, 0.f, 0.f};
#pragma unroll
  for (int i = 0; i < 2; ++i)
#pragma unroll
    for (int j = 0; j < 4; ++j) acc[i][j] = vzero;

  issue(0, 0);
  const int NT = EE / 32;
  for (int kt = 0; kt < NT; ++kt) {
    const int cur = kt & 1;
    wait_async0();
    __syncthreads();                       // tile kt resident; prev reads done
    if (kt + 1 < NT) issue(kt + 1, cur ^ 1);
    const unsigned short* a = As[cur];
    const unsigned short* bsh = Bs[cur];
#pragma unroll
    for (int fm = 0; fm < 2; ++fm) {
      v16bf afr = load_frag_a(a, wm * 32 + fm * 16, LDT, 0, lane);
#pragma unroll
      for (int fn = 0; fn < 4; ++fn) {
        v16bf bfr = load_frag_b(bsh, wn * 64 + fn * 16, LDT, 0, lane);
        acc[fm][fn] = __builtin_amdgcn_wmma_f32_16x16x32_bf16(
            false, afr, false, bfr, (short)0, acc[fm][fn], false, false);
      }
    }
  }

  const int r15 = lane & 15, hi = lane >> 4;
#pragma unroll
  for (int fm = 0; fm < 2; ++fm)
#pragma unroll
    for (int fn = 0; fn < 4; ++fn) {
      const int col = n0 + wn * 64 + fn * 16 + r15;
      const float bcol = bias[col];
#pragma unroll
      for (int r = 0; r < 8; ++r) {
        const int m = m0 + wm * 32 + fm * 16 + r + 8 * hi;
        out[(size_t)m * EE + col] = f2bf((acc[fm][fn][r] + bcol) * alpha);
      }
    }
}

// ---------------------------------------------------------------------------
// K2: per-(head, chunk) local attention: o = relu(q @ k^T) @ v   (64x64x64)
// grid = (B*H, G), block = 128 (4 waves; wave w owns rows [16w,16w+16)).
// ---------------------------------------------------------------------------
__global__ __launch_bounds__(128)
void attn_kernel(const unsigned short* __restrict__ qb,
                 const unsigned short* __restrict__ kb,
                 const unsigned short* __restrict__ vb,
                 float* __restrict__ o) {
  __shared__ __align__(16) unsigned short Qs[64 * 72];
  __shared__ __align__(16) unsigned short Ks[64 * 72];
  __shared__ __align__(16) unsigned short Vt[64 * 72];   // V transposed: Vt[e][s]
  __shared__ __align__(16) unsigned short Ps[64 * 72];   // relu(logits), bf16

  const int n = blockIdx.x;             // b*H + h
  const int g = blockIdx.y;
  const int b = n >> 4, h = n & 15;     // H = 16
  const int tid = threadIdx.x;
  const int lane = tid & 31, wave = tid >> 5;

  // each thread owns a half-row: 32 bf16 = 64 bytes = 4x async B128
  const int row = tid >> 1;
  const int half = (tid & 1) * 32;
  const size_t gidx = ((size_t)(g * 64 + row) * BB + b) * EE + h * 64 + half;

  const unsigned long long qg = (unsigned long long)(uintptr_t)(qb + gidx);
  const unsigned long long kg = (unsigned long long)(uintptr_t)(kb + gidx);
  const unsigned ql = lds_off(&Qs[row * 72 + half]);
  const unsigned kl = lds_off(&Ks[row * 72 + half]);
#pragma unroll
  for (int c = 0; c < 4; ++c) {
    async_ld16(ql + 16 * c, qg + 16 * c);
    async_ld16(kl + 16 * c, kg + 16 * c);
  }
  // V: vector-load 32 bf16, transpose-store into LDS.
  {
    const u32x4* vs = (const u32x4*)(vb + gidx);
    u32x4 w0 = vs[0], w1 = vs[1], w2 = vs[2], w3 = vs[3];
    unsigned t[16] = {w0.x, w0.y, w0.z, w0.w, w1.x, w1.y, w1.z, w1.w,
                      w2.x, w2.y, w2.z, w2.w, w3.x, w3.y, w3.z, w3.w};
#pragma unroll
    for (int j = 0; j < 16; ++j) {
      Vt[(half + 2 * j) * 72 + row]     = (unsigned short)(t[j] & 0xffffu);
      Vt[(half + 2 * j + 1) * 72 + row] = (unsigned short)(t[j] >> 16);
    }
  }
  wait_async0();
  __syncthreads();

  const v8f vzero = {0.f, 0.f, 0.f, 0.f, 0.f, 0.f, 0.f, 0.f};

  // logits = q @ k^T (K = HD = 64 -> two x32 steps)
  v8f acc[4] = {vzero, vzero, vzero, vzero};
#pragma unroll
  for (int ks = 0; ks < 64; ks += 32) {
    v16bf afr = load_frag_a(Qs, wave * 16, 72, ks, lane);
#pragma unroll
    for (int fn = 0; fn < 4; ++fn) {
      v16bf bfr = load_frag_b(Ks, fn * 16, 72, ks, lane);
      acc[fn] = __builtin_amdgcn_wmma_f32_16x16x32_bf16(
          false, afr, false, bfr, (short)0, acc[fn], false, false);
    }
  }

  // relu -> bf16 prob tile in LDS
  const int r15 = lane & 15, hi = lane >> 4;
#pragma unroll
  for (int fn = 0; fn < 4; ++fn)
#pragma unroll
    for (int r = 0; r < 8; ++r)
      Ps[(wave * 16 + r + 8 * hi) * 72 + fn * 16 + r15] =
          f2bf(fmaxf(acc[fn][r], 0.0f));
  __syncthreads();

  // o = prob @ v == prob @ (Vt)^T  (K = C = 64)
  v8f acc2[4] = {vzero, vzero, vzero, vzero};
#pragma unroll
  for (int ks = 0; ks < 64; ks += 32) {
    v16bf afr = load_frag_a(Ps, wave * 16, 72, ks, lane);
#pragma unroll
    for (int fn = 0; fn < 4; ++fn) {
      v16bf bfr = load_frag_b(Vt, fn * 16, 72, ks, lane);
      acc2[fn] = __builtin_amdgcn_wmma_f32_16x16x32_bf16(
          false, afr, false, bfr, (short)0, acc2[fn], false, false);
    }
  }

#pragma unroll
  for (int fn = 0; fn < 4; ++fn)
#pragma unroll
    for (int r = 0; r < 8; ++r) {
      const int c = wave * 16 + r + 8 * hi;
      const int col = fn * 16 + r15;
      o[((size_t)(g * 64 + c) * BB + b) * EE + h * 64 + col] = acc2[fn][r];
    }
}

// ---------------------------------------------------------------------------
// K3: gated RMS norm over E per token row; bf16 output.
// ---------------------------------------------------------------------------
__global__ __launch_bounds__(256)
void norm_kernel(const float* __restrict__ o,
                 const float* __restrict__ scale,
                 const float* __restrict__ gate,
                 unsigned short* __restrict__ onorm) {
  __shared__ float red[8];
  const int m = blockIdx.x;
  const int tid = threadIdx.x;
  const float* x = o + (size_t)m * EE;

  f32x4 v = *(const f32x4*)(x + tid * 4);
  float ss = v.x * v.x + v.y * v.y + v.z * v.z + v.w * v.w;
#pragma unroll
  for (int off = 16; off > 0; off >>= 1) ss += __shfl_xor(ss, off, 32);
  if ((tid & 31) == 0) red[tid >> 5] = ss;
  __syncthreads();
  float tot = 0.f;
#pragma unroll
  for (int i = 0; i < 8; ++i) tot += red[i];

  const float rms = sqrtf(tot) * 0.03125f;          // * E^-0.5 (E=1024)
  const float inv = 1.0f / (rms + 1e-8f);

  const int e0 = tid * 4;
  f32x4 sc = *(const f32x4*)(scale + e0);
  f32x4 gt = *(const f32x4*)(gate + e0);
  float xv[4] = {v.x, v.y, v.z, v.w};
  float sv[4] = {sc.x, sc.y, sc.z, sc.w};
  float gv[4] = {gt.x, gt.y, gt.z, gt.w};
  unsigned short r[4];
#pragma unroll
  for (int i = 0; i < 4; ++i) {
    const float sig = 1.0f / (1.0f + __expf(-gv[i] * xv[i]));
    r[i] = f2bf(sv[i] * xv[i] * inv * sig);
  }
  u32x2 p = {(unsigned)r[0] | ((unsigned)r[1] << 16),
             (unsigned)r[2] | ((unsigned)r[3] << 16)};
  *(u32x2*)(onorm + (size_t)m * EE + e0) = p;
}

// ---------------------------------------------------------------------------
// K4: output projection: out = o_normed @ Wo.T + bo (fp32), async-staged.
// ---------------------------------------------------------------------------
__global__ __launch_bounds__(256)
void out_gemm_kernel(const unsigned short* __restrict__ an,
                     const unsigned short* __restrict__ wob,
                     const float* __restrict__ bo,
                     float* __restrict__ out) {
  __shared__ __align__(16) unsigned short As[2][128 * LDT];
  __shared__ __align__(16) unsigned short Bs[2][128 * LDT];

  const int m0 = blockIdx.x * 128, n0 = blockIdx.y * 128;
  const int tid = threadIdx.x, lane = tid & 31, wave = tid >> 5;
  const int wm = wave & 3, wn = wave >> 2;

  const int srow = tid >> 1, sseg = (tid & 1) * 16;
  const unsigned long long abase =
      (unsigned long long)(uintptr_t)(an + (size_t)(m0 + srow) * EE + sseg);
  const unsigned long long bbase =
      (unsigned long long)(uintptr_t)(wob + (size_t)(n0 + srow) * EE + sseg);
  const unsigned alds[2] = {lds_off(&As[0][srow * LDT + sseg]),
                            lds_off(&As[1][srow * LDT + sseg])};
  const unsigned blds[2] = {lds_off(&Bs[0][srow * LDT + sseg]),
                            lds_off(&Bs[1][srow * LDT + sseg])};

  auto issue = [&](int kt, int buf) {
    const unsigned long long ko64 = (unsigned long long)kt * 64ull;
    async_ld16(alds[buf], abase + ko64);
    async_ld16(alds[buf] + 32, abase + ko64 + 32);
    async_ld16(blds[buf], bbase + ko64);
    async_ld16(blds[buf] + 32, bbase + ko64 + 32);
  };

  v8f acc[2][4];
  const v8f vzero = {0.f, 0.f, 0.f, 0.f, 0.f, 0.f, 0.f, 0.f};
#pragma unroll
  for (int i = 0; i < 2; ++i)
#pragma unroll
    for (int j = 0; j < 4; ++j) acc[i][j] = vzero;

  issue(0, 0);
  const int NT = EE / 32;
  for (int kt = 0; kt < NT; ++kt) {
    const int cur = kt & 1;
    wait_async0();
    __syncthreads();
    if (kt + 1 < NT) issue(kt + 1, cur ^ 1);
    const unsigned short* a = As[cur];
    const unsigned short* bsh = Bs[cur];
#pragma unroll
    for (int fm = 0; fm < 2; ++fm) {
      v16bf afr = load_frag_a(a, wm * 32 + fm * 16, LDT, 0, lane);
#pragma unroll
      for (int fn = 0; fn < 4; ++fn) {
        v16bf bfr = load_frag_b(bsh, wn * 64 + fn * 16, LDT, 0, lane);
        acc[fm][fn] = __builtin_amdgcn_wmma_f32_16x16x32_bf16(
            false, afr, false, bfr, (short)0, acc[fm][fn], false, false);
      }
    }
  }

  const int r15 = lane & 15, hi = lane >> 4;
#pragma unroll
  for (int fm = 0; fm < 2; ++fm)
#pragma unroll
    for (int fn = 0; fn < 4; ++fn) {
      const int col = n0 + wn * 64 + fn * 16 + r15;
      const float bcol = bo[col];
#pragma unroll
      for (int r = 0; r < 8; ++r) {
        const int m = m0 + wm * 32 + fm * 16 + r + 8 * hi;
        out[(size_t)m * EE + col] = acc[fm][fn][r] + bcol;
      }
    }
}

// ---------------------------------------------------------------------------
extern "C" void kernel_launch(void* const* d_in, const int* in_sizes, int n_in,
                              void* d_out, int out_size, void* d_ws, size_t ws_size,
                              hipStream_t stream) {
  (void)in_sizes; (void)n_in; (void)out_size; (void)ws_size;
  const float* query = (const float*)d_in[0];
  const float* Wq = (const float*)d_in[1];  const float* bq = (const float*)d_in[2];
  const float* Wk = (const float*)d_in[3];  const float* bk = (const float*)d_in[4];
  const float* Wv = (const float*)d_in[5];  const float* bv = (const float*)d_in[6];
  const float* Wo = (const float*)d_in[7];  const float* bo = (const float*)d_in[8];
  const float* nscale = (const float*)d_in[9];
  const float* ngate  = (const float*)d_in[10];

  const size_t S = (size_t)MM * EE;                   // 16M elements
  unsigned short* xb    = (unsigned short*)d_ws;      // bf16 x        (32 MB)
  unsigned short* wball = xb + S;                     // bf16 Wq/k/v/o ( 8 MB)
  unsigned short* qb    = wball + (size_t)4 * EE * EE;// bf16 q        (32 MB)
  unsigned short* kb    = qb + S;                     // bf16 k        (32 MB)
  unsigned short* vb    = kb + S;                     // bf16 v        (32 MB)
  float*          ob    = (float*)(vb + S);           // f32 o         (64 MB)
  unsigned short* nb    = (unsigned short*)(ob + S);  // bf16 o_normed (32 MB)

  convert_x_kernel<<<MM, 256, 0, stream>>>(query, xb);
  convert_w_kernel<<<dim3(EE, 4), 256, 0, stream>>>(Wq, Wk, Wv, Wo, wball);
  qkv_proj_kernel<<<dim3(MM / 128, EE / 128, 3), 256, 0, stream>>>(
      xb, wball, bq, bk, bv, qb, kb, vb);
  attn_kernel<<<dim3(BB * HH, GG), 128, 0, stream>>>(qb, kb, vb, ob);
  norm_kernel<<<MM, 256, 0, stream>>>(ob, nscale, ngate, nb);
  out_gemm_kernel<<<dim3(MM / 128, EE / 128), 256, 0, stream>>>(
      nb, wball + (size_t)3 * EE * EE, bo, (float*)d_out);
}
